// GCN_35287451304491
// MI455X (gfx1250) — matrix operand order repaired and checked by reading, not compiled, not run on hardware
//
#include <hip/hip_runtime.h>
#include <hip/hip_bf16.h>
#include <math.h>

typedef __attribute__((ext_vector_type(16))) __bf16 v16bf;
typedef __attribute__((ext_vector_type(8)))  float  v8f;

#define DIM_IN   500
#define DIM_H    64
#define DIM_OUT  8
#define K_TILES  16          // K padded to 512 = 16 tiles of 32
#define K_MAIN   15          // tiles with no bounds checks (k < 480)

// ---------------------------------------------------------------- utilities
__global__ void k_zero_f32(float* __restrict__ p, int n) {
    int i = blockIdx.x * blockDim.x + threadIdx.x;
    if (i < n) p[i] = 0.0f;
}

__global__ void k_fill_f32(float* __restrict__ p, float v, int n) {
    int i = blockIdx.x * blockDim.x + threadIdx.x;
    if (i < n) p[i] = v;
}

// Split W1 (f32 [500,64]) into bf16 hi/lo, stored in WMMA B-fragment order,
// K zero-padded to 512. Element index i encodes:
//   e = i & 15, half = (i>>4)&1, col = (i>>5)&63, t = i>>11
//   k = 32*t + 8*half + (e&7) + 16*(e>=8)   (B layout: N = lane&15, table K order)
// so a lane's 16 B elements are the 32 contiguous bytes at ((t*64+col)*2+half)*16.
__global__ void k_split_w1_swz(const float* __restrict__ w, __bf16* __restrict__ whi,
                               __bf16* __restrict__ wlo) {
    int i = blockIdx.x * blockDim.x + threadIdx.x;
    if (i >= K_TILES * 64 * 2 * 16) return;
    int e    = i & 15;
    int half = (i >> 4) & 1;
    int col  = (i >> 5) & 63;
    int t    = i >> 11;
    int k    = 32 * t + 8 * half + (e & 7) + ((e >= 8) ? 16 : 0);
    float v  = (k < DIM_IN) ? w[k * DIM_H + col] : 0.0f;
    __bf16 h = (__bf16)v;
    whi[i] = h;
    wlo[i] = (__bf16)(v - (float)h);
}

// deg[dst] += 1  (deg pre-initialized to 1.0 == self loop)
__global__ void k_degree(const int* __restrict__ dst, float* __restrict__ deg, int n_edges) {
    int e = blockIdx.x * blockDim.x + threadIdx.x;
    if (e < n_edges) atomicAdd(&deg[dst[e]], 1.0f);
}

__global__ void k_rsqrt_inplace(float* __restrict__ d, int n) {
    int i = blockIdx.x * blockDim.x + threadIdx.x;
    if (i < n) {
        float v = d[i];
        d[i] = (v > 0.0f) ? rsqrtf(v) : 0.0f;
    }
}

// ---------------------------------------------------------------- GEMM1 (WMMA bf16x3)
// hw[100000,64] = x[100000,500] @ W1[500,64], ~f32 accuracy via hi/lo bf16 split.
// Block = 128 threads = 4 waves; wave w owns column tile [16w,16w+16);
// blockIdx.x owns row tile [16b,16b+16). 100000/16 = 6250 exact.

__device__ __forceinline__ void make_a_frag(const float* __restrict__ va,
                                            v16bf& ahi, v16bf& alo) {
#pragma unroll
    for (int e = 0; e < 16; ++e) {
        float v  = va[e];
        __bf16 h = (__bf16)v;
        ahi[e] = h;
        alo[e] = (__bf16)(v - (float)h);
    }
}

__global__ __launch_bounds__(128) void k_gemm1_wmma(
    const float* __restrict__ x,
    const __bf16* __restrict__ w1hi,
    const __bf16* __restrict__ w1lo,
    float* __restrict__ hw) {
    const int wave = threadIdx.x >> 5;          // 0..3 -> N tile
    const int lane = threadIdx.x & 31;
    const int l16  = lane & 15;
    const int hiH  = (lane >> 4) & 1;           // 0 or 1
    const int rowT = blockIdx.x * 16;
    const int row  = rowT + l16;                // A row for this lane
    const int col  = wave * 16 + l16;           // B/C column for this lane

    const float* __restrict__ xrow = x + (size_t)row * DIM_IN;

    v8f acc = {};

    // ---- main K loop: k0 = 0..448, no bounds checks anywhere
    for (int t = 0; t < K_MAIN; ++t) {
        const int k0    = t * 32;
        const int aBase = k0 + hiH * 8;     // A: e0..7 -> aBase+e ; e8..15 -> aBase+16+(e-8)
        float va[16];
        *(float4*)(va + 0)  = *(const float4*)(xrow + aBase);
        *(float4*)(va + 4)  = *(const float4*)(xrow + aBase + 4);
        *(float4*)(va + 8)  = *(const float4*)(xrow + aBase + 16);
        *(float4*)(va + 12) = *(const float4*)(xrow + aBase + 20);
        v16bf ahi, alo;
        make_a_frag(va, ahi, alo);

        const size_t bIdx = ((size_t)(t * 64 + col) * 2 + hiH) * 16;
        v16bf bhi = *(const v16bf*)(w1hi + bIdx);
        v16bf blo = *(const v16bf*)(w1lo + bIdx);

        acc = __builtin_amdgcn_wmma_f32_16x16x32_bf16(false, ahi, false, bhi, (short)0, acc, false, false);
        acc = __builtin_amdgcn_wmma_f32_16x16x32_bf16(false, alo, false, bhi, (short)0, acc, false, false);
        acc = __builtin_amdgcn_wmma_f32_16x16x32_bf16(false, ahi, false, blo, (short)0, acc, false, false);
    }

    // ---- tail tile t = 15 (k0 = 480, valid K = 480..499)
    {
        const int t     = K_MAIN;
        const int k0    = t * 32;
        const int aBase = k0 + hiH * 8;
        float va[16];
#pragma unroll
        for (int e = 0; e < 16; ++e) {
            int k  = aBase + e + ((e >= 8) ? 8 : 0);
            int kc = (k < DIM_IN) ? k : (DIM_IN - 1);   // clamp address (in bounds)
            float v = xrow[kc];
            va[e] = (k < DIM_IN) ? v : 0.0f;            // select, no EXEC divergence
        }
        v16bf ahi, alo;
        make_a_frag(va, ahi, alo);

        const size_t bIdx = ((size_t)(t * 64 + col) * 2 + hiH) * 16;  // zero-padded
        v16bf bhi = *(const v16bf*)(w1hi + bIdx);
        v16bf blo = *(const v16bf*)(w1lo + bIdx);

        acc = __builtin_amdgcn_wmma_f32_16x16x32_bf16(false, ahi, false, bhi, (short)0, acc, false, false);
        acc = __builtin_amdgcn_wmma_f32_16x16x32_bf16(false, alo, false, bhi, (short)0, acc, false, false);
        acc = __builtin_amdgcn_wmma_f32_16x16x32_bf16(false, ahi, false, blo, (short)0, acc, false, false);
    }

    // ---- C/D layout: VGPR r -> row = rowT + r + 8*(lane>=16), col = lane&15 (+16w)
#pragma unroll
    for (int r = 0; r < 8; ++r) {
        int m = rowT + r + hiH * 8;
        hw[(size_t)m * DIM_H + col] = acc[r];
    }
}

// ---------------------------------------------------------------- edge scatter, layer 1
// 16 threads per edge, each handles 4 contiguous features (float4 gather, 4 atomics).
__global__ void k_scatter1(const int* __restrict__ src, const int* __restrict__ dst,
                           const float* __restrict__ dinv, const float* __restrict__ hw,
                           float* __restrict__ agg, int n_edges) {
    long long gid = (long long)blockIdx.x * blockDim.x + threadIdx.x;
    if (gid >= (long long)n_edges * 16) return;
    int e = (int)(gid >> 4);
    int c = ((int)gid & 15) * 4;
    int s = src[e], d = dst[e];
    float nrm = dinv[s] * dinv[d];
    const float4 v = *(const float4*)(hw + (size_t)s * DIM_H + c);
    float* o = agg + (size_t)d * DIM_H + c;
    atomicAdd(o + 0, v.x * nrm);
    atomicAdd(o + 1, v.y * nrm);
    atomicAdd(o + 2, v.z * nrm);
    atomicAdd(o + 3, v.w * nrm);
}

// agg1 <- relu(agg1 + hw1*dinv^2 + b1)   (self-loop + bias + relu; h stored in place)
__global__ void k_finalize1(float* __restrict__ agg, const float* __restrict__ hw,
                            const float* __restrict__ dinv, const float* __restrict__ b1,
                            int n) {
    int gid = blockIdx.x * blockDim.x + threadIdx.x;
    if (gid >= n) return;
    int node = gid >> 6;
    int f    = gid & 63;
    float di = dinv[node];
    float v = agg[gid] + hw[gid] * di * di + b1[f];
    agg[gid] = fmaxf(v, 0.0f);
}

// ---------------------------------------------------------------- GEMM2 (VALU; 0.1 GFLOP, N=8)
__global__ void k_gemm2(const float* __restrict__ h, const float* __restrict__ W2,
                        float* __restrict__ hw2, int n_nodes) {
    int gid = blockIdx.x * blockDim.x + threadIdx.x;
    if (gid >= n_nodes * DIM_OUT) return;
    int node = gid >> 3;
    int o    = gid & 7;
    const float* hr = h + (size_t)node * DIM_H;
    float s = 0.0f;
#pragma unroll
    for (int k = 0; k < DIM_H; ++k) s = fmaf(hr[k], W2[k * DIM_OUT + o], s);
    hw2[gid] = s;
}

// ---------------------------------------------------------------- edge scatter, layer 2
__global__ void k_scatter2(const int* __restrict__ src, const int* __restrict__ dst,
                           const float* __restrict__ dinv, const float* __restrict__ hw2,
                           float* __restrict__ agg2, int n_edges) {
    long long gid = (long long)blockIdx.x * blockDim.x + threadIdx.x;
    if (gid >= (long long)n_edges * 2) return;
    int e = (int)(gid >> 1);
    int c = ((int)gid & 1) * 4;
    int s = src[e], d = dst[e];
    float nrm = dinv[s] * dinv[d];
    const float4 v = *(const float4*)(hw2 + (size_t)s * DIM_OUT + c);
    float* o = agg2 + (size_t)d * DIM_OUT + c;
    atomicAdd(o + 0, v.x * nrm);
    atomicAdd(o + 1, v.y * nrm);
    atomicAdd(o + 2, v.z * nrm);
    atomicAdd(o + 3, v.w * nrm);
}

// z = agg2 + hw2*dinv^2 + b2 ; out = log_softmax(z) per node
__global__ void k_finalize2(const float* __restrict__ agg2, const float* __restrict__ hw2,
                            const float* __restrict__ dinv, const float* __restrict__ b2,
                            float* __restrict__ out, int n_nodes) {
    int i = blockIdx.x * blockDim.x + threadIdx.x;
    if (i >= n_nodes) return;
    float di = dinv[i];
    float sl = di * di;
    float z[DIM_OUT];
    float m = -INFINITY;
#pragma unroll
    for (int o = 0; o < DIM_OUT; ++o) {
        z[o] = agg2[(size_t)i * DIM_OUT + o] + hw2[(size_t)i * DIM_OUT + o] * sl + b2[o];
        m = fmaxf(m, z[o]);
    }
    float s = 0.0f;
#pragma unroll
    for (int o = 0; o < DIM_OUT; ++o) s += expf(z[o] - m);
    float lse = m + logf(s);
#pragma unroll
    for (int o = 0; o < DIM_OUT; ++o) out[(size_t)i * DIM_OUT + o] = z[o] - lse;
}

// ---------------------------------------------------------------- host side
static inline int cdiv(long long n, int b) { return (int)((n + b - 1) / b); }

extern "C" void kernel_launch(void* const* d_in, const int* in_sizes, int n_in,
                              void* d_out, int out_size, void* d_ws, size_t ws_size,
                              hipStream_t stream) {
    const float* x   = (const float*)d_in[0];
    const int*   ei  = (const int*)d_in[1];
    const float* W1  = (const float*)d_in[2];
    const float* b1  = (const float*)d_in[3];
    const float* W2  = (const float*)d_in[4];
    const float* b2  = (const float*)d_in[5];
    float*       out = (float*)d_out;

    const int n_nodes = in_sizes[0] / DIM_IN;      // 100000
    const int n_edges = in_sizes[1] / 2;           // 1600000
    const int* src = ei;
    const int* dst = ei + n_edges;

    // carve workspace (256B aligned)
    char* ws = (char*)d_ws;
    size_t off = 0;
    auto carve = [&](size_t bytes) -> char* {
        char* p = ws + off;
        off = (off + bytes + 255) & ~(size_t)255;
        return p;
    };
    const size_t w1swz = (size_t)K_TILES * 64 * 2 * 16;                 // 32768 elems
    float*  dinv  = (float*)carve((size_t)n_nodes * 4);                 // deg -> dinv in place
    __bf16* w1hi  = (__bf16*)carve(w1swz * 2);
    __bf16* w1lo  = (__bf16*)carve(w1swz * 2);
    float*  hw1   = (float*)carve((size_t)n_nodes * DIM_H * 4);
    float*  agg1  = (float*)carve((size_t)n_nodes * DIM_H * 4);         // becomes h in place
    float*  hw2   = (float*)carve((size_t)n_nodes * DIM_OUT * 4);
    float*  agg2  = (float*)carve((size_t)n_nodes * DIM_OUT * 4);
    (void)ws_size; (void)n_in; (void)out_size;

    const int B  = 256;
    const int nH = n_nodes * DIM_H;     // 6.4M
    const int nO = n_nodes * DIM_OUT;   // 0.8M

    // degree / normalization
    k_fill_f32<<<cdiv(n_nodes, B), B, 0, stream>>>(dinv, 1.0f, n_nodes);   // self loop
    k_split_w1_swz<<<cdiv((long long)w1swz, B), B, 0, stream>>>(W1, w1hi, w1lo);
    k_degree<<<cdiv(n_edges, B), B, 0, stream>>>(dst, dinv, n_edges);
    k_rsqrt_inplace<<<cdiv(n_nodes, B), B, 0, stream>>>(dinv, n_nodes);

    // layer 1
    k_zero_f32<<<cdiv(nH, B), B, 0, stream>>>(agg1, nH);
    k_gemm1_wmma<<<n_nodes / 16, 128, 0, stream>>>(x, w1hi, w1lo, hw1);
    k_scatter1<<<cdiv((long long)n_edges * 16, B), B, 0, stream>>>(src, dst, dinv, hw1, agg1, n_edges);
    k_finalize1<<<cdiv(nH, B), B, 0, stream>>>(agg1, hw1, dinv, b1, nH);

    // layer 2
    k_zero_f32<<<cdiv(nO, B), B, 0, stream>>>(agg2, nO);
    k_gemm2<<<cdiv(nO, B), B, 0, stream>>>(agg1, W2, hw2, n_nodes);
    k_scatter2<<<cdiv((long long)n_edges * 2, B), B, 0, stream>>>(src, dst, dinv, hw2, agg2, n_edges);
    k_finalize2<<<cdiv(n_nodes, B), B, 0, stream>>>(agg2, hw2, dinv, b2, out, n_nodes);
}
